// GATv2Convolution_88055419503316
// MI455X (gfx1250) — matrix operand order repaired and checked by reading, not compiled, not run on hardware
//
#include <hip/hip_runtime.h>

typedef __attribute__((ext_vector_type(16))) __bf16 v16bf;
typedef __attribute__((ext_vector_type(8)))  float  v8f;

#define IN_   128
#define HID_  32
#define HEADS_ 4
#define H1_   128
#define EDIM_ 16
#define C2_   8
#define OUT_  8

__device__ __forceinline__ unsigned short f2bf(float f) {
  unsigned int u = __float_as_uint(f);
  unsigned int r = u + 0x7FFFu + ((u >> 16) & 1u);   // round-to-nearest-even
  return (unsigned short)(r >> 16);
}

__device__ __forceinline__ void atomicMaxF(float* addr, float val) {
  if (val >= 0.0f) atomicMax((int*)addr, __float_as_int(val));
  else             atomicMin((unsigned int*)addr, __float_as_uint(val));
}

// ---------------------------------------------------------------------------
// Generic bf16 WMMA GEMM:  C[M,Nout] = act(A[M,K] @ W[K,Nout] + bias)
// K <= 128 (padded to 32), Nout <= 128 (padded to 16). 128 thr = 4 waves,
// 64 rows / block. W and A staged in LDS as bf16.
// act: 0 = none, 1 = ELU, 2 = sigmoid
// ---------------------------------------------------------------------------
__global__ __launch_bounds__(128) void k_gemm_bf16(
    const float* __restrict__ A, const float* __restrict__ W,
    const float* __restrict__ bias, float* __restrict__ C,
    int M, int K, int Nout, int act)
{
  __shared__ unsigned short Wlds[128 * 128];  // [n][k], 32 KB
  __shared__ unsigned short Alds[64 * 128];   // [r][k], 16 KB

  const int Kp = (K + 31) & ~31;
  const int Np = (Nout + 15) & ~15;
  const long rowBase = (long)blockIdx.x * 64;

  // stage W transposed (column-per-row in LDS), zero-padded
  for (int i = threadIdx.x; i < Np * Kp; i += blockDim.x) {
    int n = i / Kp, k = i % Kp;
    float v = (k < K && n < Nout) ? W[(long)k * Nout + n] : 0.0f;
    Wlds[i] = f2bf(v);
  }
  // stage A tile, zero-padded
  for (int i = threadIdx.x; i < 64 * Kp; i += blockDim.x) {
    int r = i / Kp, k = i % Kp;
    long row = rowBase + r;
    float v = (row < M && k < K) ? A[row * (long)K + k] : 0.0f;
    Alds[i] = f2bf(v);
  }
  __syncthreads();

  const int wave = threadIdx.x >> 5;
  const int lane = threadIdx.x & 31;
  const int g    = lane >> 4;
  const int l16  = lane & 15;
  const long wRow = rowBase + wave * 16;

  const unsigned short* arow = &Alds[(wave * 16 + l16) * Kp];

  for (int nt = 0; nt < (Np >> 4); ++nt) {
    union { v8f v; float f[8]; } acc;
#pragma unroll
    for (int r = 0; r < 8; ++r) acc.f[r] = 0.0f;

    const unsigned short* brow = &Wlds[(nt * 16 + l16) * Kp];

    for (int kt = 0; kt < Kp; kt += 32) {
      union { v16bf v; unsigned int u[8]; } a, b;
#pragma unroll
      for (int j = 0; j < 8; ++j) {
        int ka = kt + (j < 4 ? 2 * j : 8 + 2 * j) + 8 * g;  // A 16x32 layout
        int kb = kt + (j < 4 ? 2 * j : 8 + 2 * j) + 8 * g;  // B 32x16 layout
        a.u[j] = *(const unsigned int*)(arow + ka);
        b.u[j] = *(const unsigned int*)(brow + kb);
      }
      acc.v = __builtin_amdgcn_wmma_f32_16x16x32_bf16(
          false, a.v, false, b.v, (short)0, acc.v, false, false);
    }

    int col = nt * 16 + l16;
    if (col < Nout) {
      float bv = bias ? bias[col] : 0.0f;
#pragma unroll
      for (int r = 0; r < 8; ++r) {
        long row = wRow + r + 8 * g;   // f32 C/D layout: VGPR r -> M=r / M=r+8
        if (row < M) {
          float o = acc.f[r] + bv;
          if (act == 1)      o = (o > 0.0f) ? o : (expf(o) - 1.0f);
          else if (act == 2) o = 1.0f / (1.0f + expf(-o));
          C[row * (long)Nout + col] = o;
        }
      }
    }
  }
}

// ---------------------------------------------------------------------------
// init workspace
// ---------------------------------------------------------------------------
__global__ void k_init(float* deg, float* loopsum, float* agg1, float* lmax1,
                       float* denom1, float* agg2, float* lmax2, float* denom2,
                       int N)
{
  long i = (long)blockIdx.x * blockDim.x + threadIdx.x;
  if (i >= (long)N * 128) return;
  agg1[i] = 0.0f;
  if (i < (long)N * 16) loopsum[i] = 0.0f;
  if (i < (long)N * 8)  agg2[i] = 0.0f;
  if (i < (long)N * 4)  { lmax1[i] = -__builtin_inff(); denom1[i] = 0.0f; }
  if (i < N)            { deg[i] = 0.0f; lmax2[i] = -__builtin_inff(); denom2[i] = 0.0f; }
}

// degree + sum of incoming edge_attr (thread per (edge, dim))
__global__ void k_deg_loop(const int* __restrict__ dst,
                           const float* __restrict__ ea,
                           float* deg, float* loopsum, long E)
{
  long i = (long)blockIdx.x * blockDim.x + threadIdx.x;
  if (i >= E * EDIM_) return;
  long e = i >> 4;
  int  d = (int)(i & 15);
  int  t = dst[e];
  atomicAdd(&loopsum[(long)t * EDIM_ + d], ea[i]);
  if (d == 0) atomicAdd(&deg[t], 1.0f);
}

__global__ void k_loop_div(float* loopsum, const float* deg, long n16)
{
  long i = (long)blockIdx.x * blockDim.x + threadIdx.x;
  if (i >= n16) return;
  loopsum[i] /= fmaxf(deg[i >> 4], 1.0f);
}

// ---------------------------------------------------------------------------
// layer 1 edge passes (wave per edge, 4 heads x 32 channels)
// ---------------------------------------------------------------------------
__global__ __launch_bounds__(256) void k_edge_logits1(
    const int* __restrict__ src, const int* __restrict__ dst, long E, int N,
    const float* __restrict__ edge_attr, const float* __restrict__ loop_attr,
    const float* __restrict__ xl, const float* __restrict__ xr,
    const float* __restrict__ We1, const float* __restrict__ att1,
    float* __restrict__ logits, float* __restrict__ lmax)
{
  __shared__ float We[EDIM_ * H1_];
  for (int i = threadIdx.x; i < EDIM_ * H1_; i += blockDim.x) We[i] = We1[i];
  __syncthreads();

  const int wave = threadIdx.x >> 5;
  const int lane = threadIdx.x & 31;
  long e = (long)blockIdx.x * 8 + wave;
  if (e >= E + N) return;

  int s, t; const float* ea;
  if (e < E) { s = src[e]; t = dst[e]; ea = edge_attr + e * EDIM_; }
  else       { s = t = (int)(e - E);   ea = loop_attr + (e - E) * EDIM_; }

  __builtin_prefetch(&xl[(long)s * H1_ + lane], 0, 0);
  float eav = (lane < EDIM_) ? ea[lane] : 0.0f;

  float l4[HEADS_];
#pragma unroll
  for (int h = 0; h < HEADS_; ++h) {
    int ch = h * HID_ + lane;
    float ef = 0.0f;
#pragma unroll
    for (int d = 0; d < EDIM_; ++d)
      ef += __shfl(eav, d, 32) * We[d * H1_ + ch];
    float m = xl[(long)s * H1_ + ch] + xr[(long)t * H1_ + ch] + ef;
    m = (m > 0.0f) ? m : 0.2f * m;            // leaky_relu(0.2)
    float p = m * att1[ch];                    // att1 flat [h*32+c]
#pragma unroll
    for (int off = 16; off; off >>= 1) p += __shfl_xor(p, off, 32);
    l4[h] = p;
  }
  if (lane == 0) {
#pragma unroll
    for (int h = 0; h < HEADS_; ++h) {
      logits[e * HEADS_ + h] = l4[h];
      atomicMaxF(&lmax[(long)t * HEADS_ + h], l4[h]);
    }
  }
}

__global__ void k_edge_softmax1(const int* __restrict__ dst, long E, int N,
                                float* __restrict__ logits,
                                const float* __restrict__ lmax,
                                float* __restrict__ denom)
{
  long i = (long)blockIdx.x * blockDim.x + threadIdx.x;
  if (i >= (E + N) * HEADS_) return;
  long e = i >> 2; int h = (int)(i & 3);
  int t = (e < E) ? dst[e] : (int)(e - E);
  float ex = expf(logits[i] - lmax[(long)t * HEADS_ + h]);
  logits[i] = ex;
  atomicAdd(&denom[(long)t * HEADS_ + h], ex);
}

__global__ __launch_bounds__(256) void k_edge_agg1(
    const int* __restrict__ src, const int* __restrict__ dst, long E, int N,
    const float* __restrict__ xl, const float* __restrict__ logits,
    const float* __restrict__ denom, float* __restrict__ agg)
{
  const int wave = threadIdx.x >> 5;
  const int lane = threadIdx.x & 31;
  long e = (long)blockIdx.x * 8 + wave;
  if (e >= E + N) return;
  int s, t;
  if (e < E) { s = src[e]; t = dst[e]; } else { s = t = (int)(e - E); }
  __builtin_prefetch(&xl[(long)s * H1_ + lane], 0, 0);
#pragma unroll
  for (int h = 0; h < HEADS_; ++h) {
    float alpha = logits[e * HEADS_ + h] / denom[(long)t * HEADS_ + h];
    int ch = h * HID_ + lane;
    atomicAdd(&agg[(long)t * H1_ + ch], xl[(long)s * H1_ + ch] * alpha);
  }
}

// elu(agg + bias) in place
__global__ void k_elu_bias(float* a, const float* __restrict__ bias,
                           long total, int Cc)
{
  long i = (long)blockIdx.x * blockDim.x + threadIdx.x;
  if (i >= total) return;
  float v = a[i] + bias[i % Cc];
  a[i] = (v > 0.0f) ? v : (expf(v) - 1.0f);
}

// ---------------------------------------------------------------------------
// layer 2 edge passes (H=1, C=8) — thread per edge
// ---------------------------------------------------------------------------
__global__ void k_edge_logits2(
    const int* __restrict__ src, const int* __restrict__ dst, long E, int N,
    const float* __restrict__ edge_attr, const float* __restrict__ loop_attr,
    const float* __restrict__ xl, const float* __restrict__ xr,
    const float* __restrict__ We2, const float* __restrict__ att2,
    float* __restrict__ logits, float* __restrict__ lmax)
{
  long e = (long)blockIdx.x * blockDim.x + threadIdx.x;
  if (e >= E + N) return;
  int s, t; const float* ea;
  if (e < E) { s = src[e]; t = dst[e]; ea = edge_attr + e * EDIM_; }
  else       { s = t = (int)(e - E);   ea = loop_attr + (e - E) * EDIM_; }
  float er[EDIM_];
#pragma unroll
  for (int d = 0; d < EDIM_; ++d) er[d] = ea[d];
  float l = 0.0f;
#pragma unroll
  for (int c = 0; c < C2_; ++c) {
    float ef = 0.0f;
#pragma unroll
    for (int d = 0; d < EDIM_; ++d) ef += er[d] * We2[d * C2_ + c];
    float m = xl[(long)s * C2_ + c] + xr[(long)t * C2_ + c] + ef;
    m = (m > 0.0f) ? m : 0.2f * m;
    l += m * att2[c];
  }
  logits[e] = l;
  atomicMaxF(&lmax[t], l);
}

__global__ void k_edge_softmax2(const int* __restrict__ dst, long E, int N,
                                float* __restrict__ logits,
                                const float* __restrict__ lmax,
                                float* __restrict__ denom)
{
  long e = (long)blockIdx.x * blockDim.x + threadIdx.x;
  if (e >= E + N) return;
  int t = (e < E) ? dst[e] : (int)(e - E);
  float ex = expf(logits[e] - lmax[t]);
  logits[e] = ex;
  atomicAdd(&denom[t], ex);
}

__global__ void k_edge_agg2(const int* __restrict__ src,
                            const int* __restrict__ dst, long E, int N,
                            const float* __restrict__ xl,
                            const float* __restrict__ logits,
                            const float* __restrict__ denom,
                            float* __restrict__ agg)
{
  long i = (long)blockIdx.x * blockDim.x + threadIdx.x;
  if (i >= (E + N) * C2_) return;
  long e = i >> 3; int c = (int)(i & 7);
  int s, t;
  if (e < E) { s = src[e]; t = dst[e]; } else { s = t = (int)(e - E); }
  float alpha = logits[e] / denom[t];
  atomicAdd(&agg[(long)t * C2_ + c], xl[(long)s * C2_ + c] * alpha);
}

// ---------------------------------------------------------------------------
extern "C" void kernel_launch(void* const* d_in, const int* in_sizes, int n_in,
                              void* d_out, int out_size, void* d_ws, size_t ws_size,
                              hipStream_t stream)
{
  (void)n_in; (void)out_size; (void)ws_size;
  const float* x    = (const float*)d_in[0];
  const int*   ei   = (const int*)d_in[1];
  const float* ea   = (const float*)d_in[2];
  const float* W1l  = (const float*)d_in[3];
  const float* b1l  = (const float*)d_in[4];
  const float* W1r  = (const float*)d_in[5];
  const float* b1r  = (const float*)d_in[6];
  const float* We1  = (const float*)d_in[7];
  const float* att1 = (const float*)d_in[8];
  const float* bias1= (const float*)d_in[9];
  const float* W2l  = (const float*)d_in[10];
  const float* b2l  = (const float*)d_in[11];
  const float* W2r  = (const float*)d_in[12];
  const float* b2r  = (const float*)d_in[13];
  const float* We2  = (const float*)d_in[14];
  const float* att2 = (const float*)d_in[15];
  const float* bias2= (const float*)d_in[16];
  const float* Wlin = (const float*)d_in[17];
  const float* blin = (const float*)d_in[18];
  float* out = (float*)d_out;

  const int  N  = in_sizes[0] / IN_;
  const long E  = in_sizes[2] / EDIM_;
  const long E2 = E + N;
  const int* src = ei;
  const int* dst = ei + E;

  // workspace carve-up (floats)
  float* ws = (float*)d_ws;
  size_t off = 0;
  float* deg     = ws + off; off += (size_t)N;
  float* loopat  = ws + off; off += (size_t)N * EDIM_;
  float* xl1     = ws + off; off += (size_t)N * H1_;
  float* xr1     = ws + off; off += (size_t)N * H1_;
  float* agg1    = ws + off; off += (size_t)N * H1_;   // becomes h1
  float* lmax1   = ws + off; off += (size_t)N * HEADS_;
  float* denom1  = ws + off; off += (size_t)N * HEADS_;
  float* logits1 = ws + off; off += (size_t)E2 * HEADS_;
  float* xl2     = ws + off; off += (size_t)N * C2_;
  float* xr2     = ws + off; off += (size_t)N * C2_;
  float* agg2    = ws + off; off += (size_t)N * C2_;   // becomes h2
  float* lmax2   = ws + off; off += (size_t)N;
  float* denom2  = ws + off; off += (size_t)N;
  float* logits2 = ws + off; off += (size_t)E2;

  const int T = 256;
  auto cdiv = [](long a, long b) { return (int)((a + b - 1) / b); };

  // init
  k_init<<<cdiv((long)N * 128, T), T, 0, stream>>>(deg, loopat, agg1, lmax1,
                                                   denom1, agg2, lmax2, denom2, N);
  // self-loop edge_attr = mean of incoming
  k_deg_loop<<<cdiv(E * EDIM_, T), T, 0, stream>>>(dst, ea, deg, loopat, E);
  k_loop_div<<<cdiv((long)N * EDIM_, T), T, 0, stream>>>(loopat, deg, (long)N * EDIM_);

  // layer-1 node transforms (WMMA)
  k_gemm_bf16<<<cdiv(N, 64), 128, 0, stream>>>(x, W1l, b1l, xl1, N, IN_, H1_, 0);
  k_gemm_bf16<<<cdiv(N, 64), 128, 0, stream>>>(x, W1r, b1r, xr1, N, IN_, H1_, 0);

  // layer-1 attention
  k_edge_logits1<<<cdiv(E2, 8), 256, 0, stream>>>(src, dst, E, N, ea, loopat,
                                                  xl1, xr1, We1, att1, logits1, lmax1);
  k_edge_softmax1<<<cdiv(E2 * HEADS_, T), T, 0, stream>>>(dst, E, N, logits1, lmax1, denom1);
  k_edge_agg1<<<cdiv(E2, 8), 256, 0, stream>>>(src, dst, E, N, xl1, logits1, denom1, agg1);
  k_elu_bias<<<cdiv((long)N * H1_, T), T, 0, stream>>>(agg1, bias1, (long)N * H1_, H1_);

  // layer-2 node transforms (WMMA)
  k_gemm_bf16<<<cdiv(N, 64), 128, 0, stream>>>(agg1, W2l, b2l, xl2, N, H1_, C2_, 0);
  k_gemm_bf16<<<cdiv(N, 64), 128, 0, stream>>>(agg1, W2r, b2r, xr2, N, H1_, C2_, 0);

  // layer-2 attention
  k_edge_logits2<<<cdiv(E2, T), T, 0, stream>>>(src, dst, E, N, ea, loopat,
                                                xl2, xr2, We2, att2, logits2, lmax2);
  k_edge_softmax2<<<cdiv(E2, T), T, 0, stream>>>(dst, E, N, logits2, lmax2, denom2);
  k_edge_agg2<<<cdiv(E2 * C2_, T), T, 0, stream>>>(src, dst, E, N, xl2, logits2, denom2, agg2);
  k_elu_bias<<<cdiv((long)N * C2_, T), T, 0, stream>>>(agg2, bias2, (long)N * C2_, C2_);

  // final linear + sigmoid (WMMA)
  k_gemm_bf16<<<cdiv(N, 64), 128, 0, stream>>>(agg2, Wlin, blin, out, N, C2_, OUT_, 2);
}
